// AttentionLayer_47098611368344
// MI455X (gfx1250) — compile-verified, matrix operand order checked by reading, not run on hardware
//
#include <hip/hip_runtime.h>
#include <math.h>

typedef __attribute__((ext_vector_type(16))) __bf16 v16bf;
typedef __attribute__((ext_vector_type(8)))  float  v8f;
typedef __attribute__((ext_vector_type(4)))  int    v4i;

#define BDIM 4
#define SEQ  4096
#define DIM  512
#define NLAB 8921
#define LPAD 8960      // 70 * 128, zero-padded label rows for guard-free staging
#define MT 128
#define NT 128
#define KC 32
#define KSTEPS (DIM / KC)
#define RINV (1.0f / 8921.0f)

union FragU { v16bf v; uint4 q[2]; };

// ---------------- CDNA5 async memory->LDS helpers (ASYNCcnt tracked) -----------------
#if __has_builtin(__builtin_amdgcn_global_load_async_to_lds_b128)
#define HAVE_ASYNC_LDS 1
#else
#define HAVE_ASYNC_LDS 0
#endif

__device__ __forceinline__ void async_cp16(const __bf16* g, __bf16* l) {
#if HAVE_ASYNC_LDS
  // toolchain signature: (v4i* global_src, v4i* lds_dst, imm offset, imm cpol)
  __builtin_amdgcn_global_load_async_to_lds_b128((v4i*)g, (v4i*)l, 0, 0);
#else
  *(uint4*)l = *(const uint4*)g;
#endif
}

template <int N>
__device__ __forceinline__ void wait_async() {
#if HAVE_ASYNC_LDS
#if __has_builtin(__builtin_amdgcn_s_wait_asynccnt)
  __builtin_amdgcn_s_wait_asynccnt(N);
#else
  asm volatile("s_wait_asynccnt %0" ::"n"(N));
#endif
#endif
}

// ---------------- GEMM1: weights = tanh(x @ W1^T) -> bf16  (f32 sources, cvt in staging) ----
__global__ __launch_bounds__(256) void gemm1_wmma(
    const float* __restrict__ X,     // [B][S][D]
    const float* __restrict__ W1,    // [D][D]
    __bf16* __restrict__ outW)       // [B][S][D] bf16
{
  __shared__ __attribute__((aligned(16))) __bf16 ldsA[MT * KC];
  __shared__ __attribute__((aligned(16))) __bf16 ldsB[NT * KC];

  const int tid  = threadIdx.x;
  const int wave = tid >> 5, lane = tid & 31;
  const int half = lane >> 4, lr = lane & 15;
  const int wm = wave & 3, wn = wave >> 2;
  const int b  = blockIdx.z;
  const int m0 = blockIdx.y * MT;
  const int n0 = blockIdx.x * NT;

  v8f c[2][4];
  {
    v8f z = {0.f, 0.f, 0.f, 0.f, 0.f, 0.f, 0.f, 0.f};
    for (int i = 0; i < 2; ++i)
      for (int j = 0; j < 4; ++j) c[i][j] = z;
  }

  const float* Ab = X + (size_t)b * SEQ * DIM;

  for (int kk = 0; kk < DIM; kk += KC) {
    __syncthreads();
    for (int i = tid; i < MT * KC / 4; i += 256) {
      int row = i >> 3, seg = i & 7;
      float4 f = *(const float4*)&Ab[(size_t)(m0 + row) * DIM + kk + seg * 4];
      __bf16* d = &ldsA[row * KC + seg * 4];
      d[0] = (__bf16)f.x; d[1] = (__bf16)f.y; d[2] = (__bf16)f.z; d[3] = (__bf16)f.w;
    }
    for (int i = tid; i < NT * KC / 4; i += 256) {
      int row = i >> 3, seg = i & 7;
      float4 f = *(const float4*)&W1[(size_t)(n0 + row) * DIM + kk + seg * 4];
      __bf16* d = &ldsB[row * KC + seg * 4];
      d[0] = (__bf16)f.x; d[1] = (__bf16)f.y; d[2] = (__bf16)f.z; d[3] = (__bf16)f.w;
    }
    __syncthreads();

    FragU a[2], bb[4];
    for (int fm = 0; fm < 2; ++fm) {
      int row = (wm * 2 + fm) * 16 + lr;
      a[fm].q[0] = *(const uint4*)&ldsA[row * KC + half * 8];
      a[fm].q[1] = *(const uint4*)&ldsA[row * KC + 16 + half * 8];
    }
    for (int fn = 0; fn < 4; ++fn) {
      int row = (wn * 4 + fn) * 16 + lr;
      bb[fn].q[0] = *(const uint4*)&ldsB[row * KC + half * 8];
      bb[fn].q[1] = *(const uint4*)&ldsB[row * KC + 16 + half * 8];
    }
    for (int fm = 0; fm < 2; ++fm)
      for (int fn = 0; fn < 4; ++fn)
        c[fm][fn] = __builtin_amdgcn_wmma_f32_16x16x32_bf16(
            false, a[fm].v, false, bb[fn].v, (short)0, c[fm][fn], false, false);
  }

  __bf16* Wb = outW + (size_t)b * SEQ * DIM;
  for (int fm = 0; fm < 2; ++fm)
    for (int fn = 0; fn < 4; ++fn) {
      int n = n0 + (wn * 4 + fn) * 16 + lr;
      int mbase = m0 + (wm * 2 + fm) * 16 + half * 8;
      for (int r = 0; r < 8; ++r)
        Wb[(size_t)(mbase + r) * DIM + n] = (__bf16)tanhf(c[fm][fn][r]);
    }
}

// ---------------- W2 f32 -> bf16, rows [NLAB, LPAD) zero ----------------
__global__ __launch_bounds__(256) void cvt_w2(const float* __restrict__ W2,
                                              __bf16* __restrict__ dst)
{
  int i = blockIdx.x * 256 + threadIdx.x;           // one float4 group
  if (i < LPAD * DIM / 4) {
    int e = i * 4;
    int l = e / DIM;
    float4 f = {0.f, 0.f, 0.f, 0.f};
    if (l < NLAB) f = *(const float4*)&W2[e];
    __bf16* d = &dst[e];
    d[0] = (__bf16)f.x; d[1] = (__bf16)f.y; d[2] = (__bf16)f.z; d[3] = (__bf16)f.w;
  }
}

// ---------------- GEMM2: raw logits = weights @ W2^T, double-buffered async staging ----
// Stores f32 transposed into att [B][L][S].
__global__ __launch_bounds__(256) void gemm2_wmma(
    const __bf16* __restrict__ Wt,   // [B][S][D] bf16
    const __bf16* __restrict__ W2b,  // [LPAD][D] bf16 (zero-padded)
    float* __restrict__ outAtt)      // [B][NLAB][S]
{
  __shared__ __attribute__((aligned(16))) __bf16 ldsA[2][MT * KC];  // 2 x 8 KB
  __shared__ __attribute__((aligned(16))) __bf16 ldsB[2][NT * KC];  // 2 x 8 KB

  const int tid  = threadIdx.x;
  const int wave = tid >> 5, lane = tid & 31;
  const int half = lane >> 4, lr = lane & 15;
  const int wm = wave & 3, wn = wave >> 2;
  const int b  = blockIdx.z;
  const int m0 = blockIdx.y * MT;
  const int n0 = blockIdx.x * NT;

  // Staging: 16B chunks. Tile = 128 rows x 4 segs = 512 chunks; thread handles
  // chunk tid (rows 0..63) and tid+256 (rows 64..127).
  const int r0 = tid >> 2, seg = tid & 3;
  const int r1 = r0 + 64;
  const __bf16* Ab = Wt + (size_t)b * SEQ * DIM;
  const __bf16* gA0 = Ab  + (size_t)(m0 + r0) * DIM + seg * 8;
  const __bf16* gA1 = Ab  + (size_t)(m0 + r1) * DIM + seg * 8;
  const __bf16* gB0 = W2b + (size_t)(n0 + r0) * DIM + seg * 8;
  const __bf16* gB1 = W2b + (size_t)(n0 + r1) * DIM + seg * 8;
  const int lo0 = r0 * KC + seg * 8;
  const int lo1 = r1 * KC + seg * 8;

  v8f c[2][4];
  {
    v8f z = {0.f, 0.f, 0.f, 0.f, 0.f, 0.f, 0.f, 0.f};
    for (int i = 0; i < 2; ++i)
      for (int j = 0; j < 4; ++j) c[i][j] = z;
  }

  // prologue: fill buffer 0 (4 async issues per wave, in-order completion)
  async_cp16(gA0, &ldsA[0][lo0]);
  async_cp16(gA1, &ldsA[0][lo1]);
  async_cp16(gB0, &ldsB[0][lo0]);
  async_cp16(gB1, &ldsB[0][lo1]);

  for (int step = 0; step < KSTEPS; ++step) {
    const int buf = step & 1;
    if (step + 1 < KSTEPS) {
      const int kk = (step + 1) * KC;
      async_cp16(gA0 + kk, &ldsA[buf ^ 1][lo0]);
      async_cp16(gA1 + kk, &ldsA[buf ^ 1][lo1]);
      async_cp16(gB0 + kk, &ldsB[buf ^ 1][lo0]);
      async_cp16(gB1 + kk, &ldsB[buf ^ 1][lo1]);
      wait_async<4>();   // the 4 copies into `buf` are complete (in-order)
    } else {
      wait_async<0>();
    }
    __syncthreads();     // all waves' copies into `buf` visible

    FragU a[2], bb[4];
    for (int fm = 0; fm < 2; ++fm) {
      int row = (wm * 2 + fm) * 16 + lr;
      a[fm].q[0] = *(const uint4*)&ldsA[buf][row * KC + half * 8];
      a[fm].q[1] = *(const uint4*)&ldsA[buf][row * KC + 16 + half * 8];
    }
    for (int fn = 0; fn < 4; ++fn) {
      int row = (wn * 4 + fn) * 16 + lr;
      bb[fn].q[0] = *(const uint4*)&ldsB[buf][row * KC + half * 8];
      bb[fn].q[1] = *(const uint4*)&ldsB[buf][row * KC + 16 + half * 8];
    }
    for (int fm = 0; fm < 2; ++fm)
      for (int fn = 0; fn < 4; ++fn)
        c[fm][fn] = __builtin_amdgcn_wmma_f32_16x16x32_bf16(
            false, a[fm].v, false, bb[fn].v, (short)0, c[fm][fn], false, false);

    __syncthreads();     // everyone done reading `buf` before it is re-filled
  }

  for (int fm = 0; fm < 2; ++fm)
    for (int fn = 0; fn < 4; ++fn) {
      int l = n0 + (wn * 4 + fn) * 16 + lr;
      if (l < NLAB) {
        int sbase = m0 + (wm * 2 + fm) * 16 + half * 8;
        float* p = outAtt + ((size_t)b * NLAB + l) * SEQ + sbase;
        float4 lo = {c[fm][fn][0], c[fm][fn][1], c[fm][fn][2], c[fm][fn][3]};
        float4 hi = {c[fm][fn][4], c[fm][fn][5], c[fm][fn][6], c[fm][fn][7]};
        *(float4*)p = lo;
        *(float4*)(p + 4) = hi;
      }
    }
}

// ---------------- per-(b,l) softmax stats over S ----------------
__global__ __launch_bounds__(256) void softmax_stats(const float* __restrict__ att,
                                                     float2* __restrict__ stats)
{
  int row = blockIdx.x;  // b*NLAB + l
  const float* p = att + (size_t)row * SEQ;
  int t = threadIdx.x;
  float vals[SEQ / 256];
  float mx = -INFINITY;
  for (int i = 0; i < SEQ / 256; ++i) { vals[i] = p[t + i * 256]; mx = fmaxf(mx, vals[i]); }
  __shared__ float red[256];
  red[t] = mx; __syncthreads();
  for (int off = 128; off > 0; off >>= 1) {
    if (t < off) red[t] = fmaxf(red[t], red[t + off]);
    __syncthreads();
  }
  mx = red[0]; __syncthreads();
  float sum = 0.f;
  for (int i = 0; i < SEQ / 256; ++i) sum += __expf(vals[i] - mx);
  red[t] = sum; __syncthreads();
  for (int off = 128; off > 0; off >>= 1) {
    if (t < off) red[t] += red[t + off];
    __syncthreads();
  }
  if (t == 0) stats[row] = make_float2(mx, 1.0f / red[0]);
}

// ---------------- normalize in place + column sums g[b][s] = sum_l att ----------------
__global__ __launch_bounds__(256) void normalize_colsum(float* __restrict__ att,
                                                        const float2* __restrict__ stats,
                                                        float* __restrict__ g)
{
  int b  = blockIdx.z;
  int l0 = blockIdx.y * 64;
  int s  = blockIdx.x * 256 + threadIdx.x;
  int lend = (l0 + 64 < NLAB) ? l0 + 64 : NLAB;
  float acc = 0.f;
  for (int l = l0; l < lend; ++l) {
    float2 st = stats[b * NLAB + l];
    size_t idx = ((size_t)b * NLAB + l) * SEQ + s;
    float v = __expf(att[idx] - st.x) * st.y;
    att[idx] = v;
    acc += v;
  }
  atomicAdd(&g[b * SEQ + s], acc);
}

// ---------------- out[b][d] = (1/R) * sum_s g[b][s] * x[b][s][d] ----------------
__global__ __launch_bounds__(256) void final_out(const float* __restrict__ x,
                                                 const float* __restrict__ g,
                                                 float* __restrict__ out)
{
  int b = blockIdx.y;
  int d = blockIdx.x * 256 + threadIdx.x;
  const float* xb = x + (size_t)b * SEQ * DIM;
  const float* gb = g + b * SEQ;
  float acc = 0.f;
  for (int s = 0; s < SEQ; ++s) acc += gb[s] * xb[(size_t)s * DIM + d];
  out[b * DIM + d] = acc * RINV;
}

extern "C" void kernel_launch(void* const* d_in, const int* in_sizes, int n_in,
                              void* d_out, int out_size, void* d_ws, size_t ws_size,
                              hipStream_t stream)
{
  const float* x  = (const float*)d_in[0];  // [B][S][D]
  const float* W1 = (const float*)d_in[1];  // [D][D]
  const float* W2 = (const float*)d_in[2];  // [L][D]
  float* out = (float*)d_out;               // [B][D]
  float* att = out + BDIM * DIM;            // [B][L][S]

  char* ws = (char*)d_ws;
  __bf16* weights = (__bf16*)ws;                       // 16.8 MB
  size_t off = (size_t)BDIM * SEQ * DIM * sizeof(__bf16);
  __bf16* w2b = (__bf16*)(ws + off);                   // 9.2 MB
  off += (size_t)LPAD * DIM * sizeof(__bf16);
  float2* stats = (float2*)(ws + off);                 // [B*L]
  off += (size_t)BDIM * NLAB * sizeof(float2);
  float* g = (float*)(ws + off);                       // [B][S]

  dim3 blk(256);

  // 0) W2 -> bf16, zero-padded to LPAD rows
  cvt_w2<<<dim3(LPAD * DIM / 4 / 256), blk, 0, stream>>>(W2, w2b);

  // 1) weights = tanh(x @ W1^T) -> bf16
  gemm1_wmma<<<dim3(DIM / NT, SEQ / MT, BDIM), blk, 0, stream>>>(x, W1, weights);

  // 2) raw logits -> att region of d_out (transposed [B][L][S])
  gemm2_wmma<<<dim3(LPAD / NT, SEQ / MT, BDIM), blk, 0, stream>>>(weights, w2b, att);

  // 3) per-(b,l) softmax stats over S
  softmax_stats<<<dim3(BDIM * NLAB), blk, 0, stream>>>(att, stats);

  // 4) g = 0, normalize att in place + column sums into g
  (void)hipMemsetAsync(g, 0, (size_t)BDIM * SEQ * sizeof(float), stream);
  normalize_colsum<<<dim3(SEQ / 256, (NLAB + 63) / 64, BDIM), blk, 0, stream>>>(att, stats, g);

  // 5) out = (1/R) * g @ x
  final_out<<<dim3(DIM / 256, BDIM), blk, 0, stream>>>(x, g, out);
}
